// WinogradConv2d_36352603193465
// MI455X (gfx1250) — matrix operand order repaired and checked by reading, not compiled, not run on hardware
//
#include <hip/hip_runtime.h>
#include <hip/hip_bf16.h>

typedef __attribute__((ext_vector_type(16))) __bf16 v16bf;
typedef __attribute__((ext_vector_type(8)))  float  v8f;

// Problem constants (fixed by reference setup_inputs)
#define BATCH 32
#define CIN   128
#define KOUT  256
#define HIN   64
#define TH    31          // tiles per dim
#define NTPI  961         // 31*31 tiles per image
#define NTILES 30752      // 32*961 ; == 961 blocks of 32 exactly
#define HOUT  62

__device__ __forceinline__ unsigned short f2bf(float f) {
  unsigned u = __float_as_uint(f);
  u += 0x7fffu + ((u >> 16) & 1u);          // round-to-nearest-even
  return (unsigned short)(u >> 16);
}
__device__ __forceinline__ float bf2f(unsigned short h) {
  return __uint_as_float(((unsigned)h) << 16);
}

// ---------------------------------------------------------------------------
// Kernel 1: U = G g G^T, split into bf16 hi/lo, pre-swizzled into the WMMA
// 16x32 bf16 A-fragment per-lane layout:
//   element (m,k): lane = m + 16*((k>>3)&1), vgpr r = (k>>4)*4 + ((k&7)>>1),
//   halfword = k&1.   Block = [pos][cchunk][kblk] of 32 lanes x 8 dwords.
// Uhi: 256KB*4 = 1MB ; Ulo at +1MB. Total workspace use: 2MB.
// ---------------------------------------------------------------------------
__global__ void wino_filter_xform(const float* __restrict__ w,
                                  unsigned int* __restrict__ Uhi,
                                  unsigned int* __restrict__ Ulo) {
  int gid = blockIdx.x * 256 + threadIdx.x;   // 16384 threads: 256 k x 64 cpairs
  int k  = gid >> 6;
  int cp = gid & 63;
  int c0 = cp * 2;

  float U0[16], U1[16];
  #pragma unroll
  for (int s = 0; s < 2; ++s) {
    const float* g = w + ((size_t)k * CIN + (c0 + s)) * 9;
    float gg[9];
    #pragma unroll
    for (int i = 0; i < 9; ++i) gg[i] = g[i];
    float t[4][3];
    #pragma unroll
    for (int j = 0; j < 3; ++j) {
      t[0][j] = gg[j];
      t[1][j] = 0.5f * (gg[j] + gg[3 + j] + gg[6 + j]);
      t[2][j] = 0.5f * (gg[j] - gg[3 + j] + gg[6 + j]);
      t[3][j] = gg[6 + j];
    }
    float* U = s ? U1 : U0;
    #pragma unroll
    for (int a = 0; a < 4; ++a) {
      U[a * 4 + 0] = t[a][0];
      U[a * 4 + 1] = 0.5f * (t[a][0] + t[a][1] + t[a][2]);
      U[a * 4 + 2] = 0.5f * (t[a][0] - t[a][1] + t[a][2]);
      U[a * 4 + 3] = t[a][2];
    }
  }

  int ckl    = c0 & 31;        // k-index within 32-chunk (even)
  int cchunk = c0 >> 5;
  int kblk   = k >> 4;
  int kl     = k & 15;         // M row
  int gsel   = ckl >> 3;
  int lane   = kl + 16 * (gsel & 1);
  int rr     = (gsel >> 1) * 4 + ((ckl & 7) >> 1);

  #pragma unroll
  for (int p = 0; p < 16; ++p) {
    unsigned short h0 = f2bf(U0[p]);
    unsigned short h1 = f2bf(U1[p]);
    unsigned short l0 = f2bf(U0[p] - bf2f(h0));
    unsigned short l1 = f2bf(U1[p] - bf2f(h1));
    unsigned idx = (((unsigned)(p * 4 + cchunk) * 16 + kblk) * 32 + lane) * 8 + rr;
    Uhi[idx] = (unsigned)h0 | ((unsigned)h1 << 16);
    Ulo[idx] = (unsigned)l0 | ((unsigned)l1 << 16);
  }
}

// ---------------------------------------------------------------------------
// Kernel 2: fused input transform + 16x bf16x3 WMMA GEMM + inverse transform.
// Workgroup: 256 threads = 8 waves = 2 tile-groups x 4 k-groups.
// Block tile: 32 Winograd tiles x 64 kout, C-loop in chunks of 32.
// A = U (M=kout), B = V (N=tile)  =>  D lanes stride over tiles => coalesced
// float2 output stores.
// ---------------------------------------------------------------------------
__global__ __launch_bounds__(256, 1)
void wino_main(const float* __restrict__ x,
               const float* __restrict__ bias,
               const unsigned int* __restrict__ Uhi_u,
               float* __restrict__ out) {
  // B-fragment staging: [pos16][ngroup2][lane32][vgpr8] dwords, hi & lo
  __shared__ __attribute__((aligned(32))) unsigned short s_hi[16 * 2 * 32 * 8 * 2];
  __shared__ __attribute__((aligned(32))) unsigned short s_lo[16 * 2 * 32 * 8 * 2];

  const int tid  = threadIdx.x;
  const int wid  = tid >> 5;
  const int lane = tid & 31;
  const int kq   = wid & 3;      // which 16-wide kout slice of the 64
  const int ngw  = wid >> 2;     // which 16-tile group of the 32
  const int kblkg = blockIdx.y * 4 + kq;     // global 16-k block (0..15)
  const int kbase = kblkg * 16;
  const int tblock = blockIdx.x * 32;        // first tile of this block

  const v16bf* Uhi = (const v16bf*)Uhi_u;
  const v16bf* Ulo = Uhi + 16 * 4 * 16 * 32; // +1MB in v16bf units

  v8f acc[16];
  #pragma unroll
  for (int p = 0; p < 16; ++p) acc[p] = (v8f){0.f,0.f,0.f,0.f,0.f,0.f,0.f,0.f};

  for (int cc = 0; cc < 4; ++cc) {
    // ---- stage V = B^T d B for 32 tiles x 32 channels into LDS ----
    #pragma unroll
    for (int it = 0; it < 4; ++it) {
      int pair = tid + it * 256;            // 1024 (tile,channel) pairs
      int t  = pair >> 5;
      int ck = pair & 31;
      int tg  = tblock + t;
      int b   = tg / NTPI;
      int rem = tg - b * NTPI;
      int tp  = rem / TH;
      int tq  = rem - tp * TH;
      const float* src = x + (((size_t)b * CIN + (cc * 32 + ck)) * HIN + tp * 2) * HIN + tq * 2;
      float d[4][4];
      #pragma unroll
      for (int i = 0; i < 4; ++i) {
        float2 a2 = *(const float2*)(src + i * HIN);
        float2 b2 = *(const float2*)(src + i * HIN + 2);
        d[i][0] = a2.x; d[i][1] = a2.y; d[i][2] = b2.x; d[i][3] = b2.y;
      }
      float tr[4][4];
      #pragma unroll
      for (int j = 0; j < 4; ++j) {
        tr[0][j] = d[0][j] - d[2][j];
        tr[1][j] = d[1][j] + d[2][j];
        tr[2][j] = d[2][j] - d[1][j];
        tr[3][j] = d[1][j] - d[3][j];
      }
      int ng = t >> 4, n = t & 15;
      int ln = n + (((ck >> 3) & 1) << 4);  // lane = n + 16*((ck>>3)&1)
      int rr = (ck >> 4) * 4 + ((ck & 7) >> 1);
      int hf = ck & 1;
      #pragma unroll
      for (int a = 0; a < 4; ++a) {
        float V0 = tr[a][0] - tr[a][2];
        float V1 = tr[a][1] + tr[a][2];
        float V2 = tr[a][2] - tr[a][1];
        float V3 = tr[a][1] - tr[a][3];
        float Vr[4] = {V0, V1, V2, V3};
        #pragma unroll
        for (int dd = 0; dd < 4; ++dd) {
          int p = a * 4 + dd;
          unsigned idx = (((unsigned)(p * 2 + ng) * 32 + ln) * 8 + rr) * 2 + hf;
          unsigned short h = f2bf(Vr[dd]);
          s_hi[idx] = h;
          s_lo[idx] = f2bf(Vr[dd] - bf2f(h));
        }
      }
    }
    __syncthreads();

    // ---- 16 positions x (hi*hi + hi*lo + lo*hi) WMMAs ----
    const v16bf* Bh = (const v16bf*)s_hi;
    const v16bf* Bl = (const v16bf*)s_lo;
    #pragma unroll
    for (int p = 0; p < 16; ++p) {
      unsigned aidx = ((unsigned)(p * 4 + cc) * 16 + kblkg) * 32 + lane;
      v16bf a_hi = Uhi[aidx];
      v16bf a_lo = Ulo[aidx];
      unsigned bidx = (unsigned)(p * 2 + ngw) * 32 + lane;
      v16bf b_hi = Bh[bidx];
      v16bf b_lo = Bl[bidx];
      acc[p] = __builtin_amdgcn_wmma_f32_16x16x32_bf16(false, a_hi, false, b_lo,
                                                       (short)0, acc[p], false, false);
      acc[p] = __builtin_amdgcn_wmma_f32_16x16x32_bf16(false, a_lo, false, b_hi,
                                                       (short)0, acc[p], false, false);
      acc[p] = __builtin_amdgcn_wmma_f32_16x16x32_bf16(false, a_hi, false, b_hi,
                                                       (short)0, acc[p], false, false);
    }
    __syncthreads();
  }

  // ---- epilogue: Y = A^T M A per lane (no shuffles), + bias, coalesced ----
  int n   = lane & 15;
  int tg  = tblock + ngw * 16 + n;
  int b   = tg / NTPI;
  int rem = tg - b * NTPI;
  int tp  = rem / TH;
  int tq  = rem - tp * TH;
  int mofs = (lane >> 4) * 8;   // lanes 16-31 hold M = v+8

  #pragma unroll
  for (int v = 0; v < 8; ++v) {
    int kout = kbase + v + mofs;
    float bk = bias[kout];
    float mm[16];
    #pragma unroll
    for (int p = 0; p < 16; ++p) mm[p] = acc[p][v];
    float t0[4], t1[4];
    #pragma unroll
    for (int dd = 0; dd < 4; ++dd) {
      t0[dd] = mm[dd] + mm[4 + dd] + mm[8 + dd];
      t1[dd] = mm[4 + dd] - mm[8 + dd] - mm[12 + dd];
    }
    float y00 = t0[0] + t0[1] + t0[2] + bk;
    float y01 = t0[1] - t0[2] - t0[3] + bk;
    float y10 = t1[0] + t1[1] + t1[2] + bk;
    float y11 = t1[1] - t1[2] - t1[3] + bk;
    float* o = out + (((size_t)b * KOUT + kout) * HOUT + tp * 2) * HOUT + tq * 2;
    *(float2*)(o)        = make_float2(y00, y01);
    *(float2*)(o + HOUT) = make_float2(y10, y11);
  }
}

extern "C" void kernel_launch(void* const* d_in, const int* in_sizes, int n_in,
                              void* d_out, int out_size, void* d_ws, size_t ws_size,
                              hipStream_t stream) {
  const float* x      = (const float*)d_in[0];
  const float* weight = (const float*)d_in[1];
  const float* bias   = (const float*)d_in[2];
  float* out = (float*)d_out;

  unsigned int* Uhi = (unsigned int*)d_ws;            // 1 MB
  unsigned int* Ulo = Uhi + (16 * 4 * 16 * 32 * 8);   // +1 MB

  // Kernel 1: 256 kout x 64 channel-pairs = 16384 threads
  wino_filter_xform<<<64, 256, 0, stream>>>(weight, Uhi, Ulo);

  // Kernel 2: 961 tile-blocks x 4 kout-blocks, 256 threads (8 waves) each
  dim3 grid(NTILES / 32, KOUT / 64);
  wino_main<<<grid, 256, 0, stream>>>(x, bias, Uhi, out);

  (void)in_sizes; (void)n_in; (void)out_size; (void)ws_size;
}
